// FlaxGPTNeoXBlock_15831249453565
// MI455X (gfx1250) — matrix operand order repaired and checked by reading, not compiled, not run on hardware
//
#include <hip/hip_runtime.h>

typedef _Float16 f16;
typedef __attribute__((ext_vector_type(16))) _Float16 v16h;
typedef __attribute__((ext_vector_type(8)))  _Float16 v8h;
typedef __attribute__((ext_vector_type(8)))  float    v8f;
typedef int v4i32 __attribute__((__vector_size__(16)));

union F16x16 { v16h v; v8h h[2]; };
union F32x8  { v8f  v; float f[8]; };

#define HID   2048
#define HEADS 16
#define HEADD 128
#define INTER 8192
#define SEQ   2048
#define BATCH 2
#define ROWS  (BATCH*SEQ)   // 4096

// ---- CDNA5 async global->LDS path (probe via __has_builtin; safe fallback) ----
#if defined(__has_builtin)
#  if __has_builtin(__builtin_amdgcn_global_load_async_to_lds_b128)
#    define HAVE_ASYNC_LDS 1
#  endif
#endif
#ifndef HAVE_ASYNC_LDS
#  define HAVE_ASYNC_LDS 0
#endif

#define GLOBAL_V4(p) ((__attribute__((address_space(1))) v4i32*)(p))
#define LOCAL_V4(p)  ((__attribute__((address_space(3))) v4i32*)(p))

#if defined(__has_builtin)
#  if __has_builtin(__builtin_amdgcn_s_wait_asynccnt)
#    define WAIT_ASYNC0() __builtin_amdgcn_s_wait_asynccnt(0)
#  endif
#endif
#ifndef WAIT_ASYNC0
#  define WAIT_ASYNC0() __asm__ volatile("s_wait_asynccnt 0" ::: "memory")
#endif

__device__ __forceinline__ float gelu_exact(float x){
  return 0.5f * x * (1.0f + erff(x * 0.7071067811865475f));
}

// ---------------------------------------------------------------- fp32 -> f16
__global__ __launch_bounds__(256) void cvt_f16_kernel(
    const float* __restrict__ in, f16* __restrict__ out, int n)
{
  int i = blockIdx.x * blockDim.x + threadIdx.x;
  int stride = gridDim.x * blockDim.x;
  for (; i < n; i += stride) out[i] = (f16)in[i];
}

// --------------------------------------------------- dual LayerNorm -> f16 x2
__global__ __launch_bounds__(256) void ln_dual_kernel(
    const float* __restrict__ hsrc,
    const float* __restrict__ s1, const float* __restrict__ b1,
    const float* __restrict__ s2, const float* __restrict__ b2,
    f16* __restrict__ x1, f16* __restrict__ x2)
{
  __shared__ float red[256];
  __shared__ float red2[256];
  int row = blockIdx.x;
  int tid = threadIdx.x;
  const float* x = hsrc + (size_t)row * HID;
  float s = 0.0f, sq = 0.0f;
  for (int i = tid; i < HID; i += 256){ float v = x[i]; s += v; sq += v*v; }
  red[tid] = s; red2[tid] = sq;
  __syncthreads();
  for (int o = 128; o > 0; o >>= 1){
    if (tid < o){ red[tid] += red[tid+o]; red2[tid] += red2[tid+o]; }
    __syncthreads();
  }
  float mu  = red[0]  * (1.0f / HID);
  float var = red2[0] * (1.0f / HID) - mu * mu;
  float rs  = rsqrtf(var + 1e-5f);
  for (int i = tid; i < HID; i += 256){
    float v = (x[i] - mu) * rs;
    x1[(size_t)row * HID + i] = (f16)(v * s1[i] + b1[i]);
    x2[(size_t)row * HID + i] = (f16)(v * s2[i] + b2[i]);
  }
}

// ------------------------------------------------- RoPE + pack Q,K,[V^T] f16
__global__ __launch_bounds__(256) void rope_pack_kernel(
    const float* __restrict__ qkv,
    f16* __restrict__ q16, f16* __restrict__ k16, f16* __restrict__ vt16)
{
  int bs = blockIdx.x;
  int b = bs / SEQ;
  int s = bs % SEQ;
  int tid = threadIdx.x;
  const float* base = qkv + (size_t)bs * (3 * HID);
  const float kln = 9.210340371976184f / 128.0f; // ln(10000)/HEADD
  for (int p = tid; p < HEADS * (HEADD/2); p += 256){
    int hh = p >> 6;
    int i  = p & 63;
    float inv = __expf(-(float)(2*i) * kln);
    float ang = (float)s * inv;
    float c = cosf(ang), sn = sinf(ang);
    const float* qp = base + hh*HEADD + 2*i;
    const float* kp = base + HID + hh*HEADD + 2*i;
    float q0 = qp[0], q1 = qp[1];
    float k0 = kp[0], k1 = kp[1];
    size_t o = ((size_t)(b*HEADS + hh) * SEQ + s) * HEADD + 2*i;
    q16[o]   = (f16)(q0*c - q1*sn);
    q16[o+1] = (f16)(q0*sn + q1*c);
    k16[o]   = (f16)(k0*c - k1*sn);
    k16[o+1] = (f16)(k0*sn + k1*c);
  }
  for (int e = tid; e < HID; e += 256){
    int hh = e >> 7;
    int d  = e & 127;
    float v = base[2*HID + e];
    vt16[((size_t)(b*HEADS + hh) * HEADD + d) * SEQ + s] = (f16)v;
  }
}

// -------------------------------------------------------- flash attention
__global__ __launch_bounds__(256) void attn_kernel(
    const f16* __restrict__ Q, const f16* __restrict__ K,
    const f16* __restrict__ Vt, const float* __restrict__ amask,
    f16* __restrict__ ao)
{
  __shared__ f16 plds[8][16*32];
  int tid = threadIdx.x;
  int wid = tid >> 5;
  int lane = tid & 31;
  int lane16 = lane & 15;
  int hi = lane >> 4;
  int gwave = blockIdx.x * 8 + wid;
  int qt = gwave & (SEQ/16 - 1);
  int bh = gwave >> 7;
  int hh = bh & (HEADS - 1);
  int b  = bh >> 4;
  int qbase = qt * 16;

  const f16* Qh = Q  + (size_t)bh * SEQ * HEADD;
  const f16* Kh = K  + (size_t)bh * SEQ * HEADD;
  const f16* Vh = Vt + (size_t)bh * HEADD * SEQ;

  // preload Q A-fragments (4 chunks of K=32 along d)
  F16x16 qa[4];
#pragma unroll
  for (int c = 0; c < 4; ++c){
    const f16* p = Qh + (size_t)(qbase + lane16) * HEADD + c*32 + hi*8;
    qa[c].h[0] = *(const v8h*)p;
    qa[c].h[1] = *(const v8h*)(p + 16);
  }

  F32x8 O[8];
  float mrow[8], lrow[8];
#pragma unroll
  for (int t = 0; t < 8; ++t)
#pragma unroll
    for (int r = 0; r < 8; ++r) O[t].f[r] = 0.0f;
#pragma unroll
  for (int r = 0; r < 8; ++r){ mrow[r] = -1e30f; lrow[r] = 0.0f; }

  const float fac = 11.313708498984761f; // sqrt(HEADD), per reference
  int nk = (qbase >> 5) + 1;
  for (int kb = 0; kb < nk; ++kb){
    int kb0 = kb * 32;
    F32x8 s0, s1;
#pragma unroll
    for (int r = 0; r < 8; ++r){ s0.f[r] = 0.0f; s1.f[r] = 0.0f; }
#pragma unroll
    for (int c = 0; c < 4; ++c){
      F16x16 kf0, kf1;
      const f16* p0 = Kh + (size_t)(kb0 + lane16) * HEADD + c*32 + hi*8;
      kf0.h[0] = *(const v8h*)p0;
      kf0.h[1] = *(const v8h*)(p0 + 16);
      const f16* p1 = Kh + (size_t)(kb0 + 16 + lane16) * HEADD + c*32 + hi*8;
      kf1.h[0] = *(const v8h*)p1;
      kf1.h[1] = *(const v8h*)(p1 + 16);
      s0.v = __builtin_amdgcn_wmma_f32_16x16x32_f16(false, qa[c].v, false, kf0.v, (short)0, s0.v, false, false);
      s1.v = __builtin_amdgcn_wmma_f32_16x16x32_f16(false, qa[c].v, false, kf1.v, (short)0, s1.v, false, false);
    }
    int k0 = kb0 + lane16;
    int k1 = kb0 + 16 + lane16;
    float am0 = amask[b * SEQ + k0];
    float am1 = amask[b * SEQ + k1];
#pragma unroll
    for (int r = 0; r < 8; ++r){
      int qi = qbase + r + hi*8;
      float a0 = s0.f[r] * fac + am0 + (k0 <= qi ? 0.0f : -1e30f);
      float a1 = s1.f[r] * fac + am1 + (k1 <= qi ? 0.0f : -1e30f);
      float mx = fmaxf(a0, a1);
#pragma unroll
      for (int off = 8; off > 0; off >>= 1)
        mx = fmaxf(mx, __shfl_xor(mx, off, 32));
      float nm = fmaxf(mrow[r], mx);
      float sc = __expf(mrow[r] - nm);
      mrow[r] = nm;
      float p0 = __expf(a0 - nm);
      float p1 = __expf(a1 - nm);
      float rs = p0 + p1;
#pragma unroll
      for (int off = 8; off > 0; off >>= 1)
        rs += __shfl_xor(rs, off, 32);
      lrow[r] = lrow[r] * sc + rs;
#pragma unroll
      for (int t = 0; t < 8; ++t) O[t].f[r] *= sc;
      s0.f[r] = p0; s1.f[r] = p1;
    }
    // C-layout -> A-fragment layout via wave-private LDS slab
    f16* pw = &plds[wid][0];
#pragma unroll
    for (int r = 0; r < 8; ++r){
      pw[(r + hi*8) * 32 + lane16]      = (f16)s0.f[r];
      pw[(r + hi*8) * 32 + 16 + lane16] = (f16)s1.f[r];
    }
    __asm__ volatile("" ::: "memory");   // keep ds_store before ds_load (same wave: HW in-order)
    F16x16 pa;
    pa.h[0] = *(const v8h*)&pw[lane16 * 32 + hi*8];
    pa.h[1] = *(const v8h*)&pw[lane16 * 32 + 16 + hi*8];
    __asm__ volatile("" ::: "memory");
#pragma unroll
    for (int t = 0; t < 8; ++t){
      F16x16 vb;
      const f16* p = Vh + (size_t)(t*16 + lane16) * SEQ + kb0 + hi*8;
      vb.h[0] = *(const v8h*)p;
      vb.h[1] = *(const v8h*)(p + 16);
      O[t].v = __builtin_amdgcn_wmma_f32_16x16x32_f16(false, pa.v, false, vb.v, (short)0, O[t].v, false, false);
    }
  }
#pragma unroll
  for (int t = 0; t < 8; ++t){
#pragma unroll
    for (int r = 0; r < 8; ++r){
      int qi = qbase + r + hi*8;
      float val = O[t].f[r] / lrow[r];
      ao[(size_t)(b * SEQ + qi) * HID + hh * HEADD + t*16 + lane16] = (f16)val;
    }
  }
}

// ------------------------------------------------ generic WMMA f16 GEMM
// Double-buffered pipeline; A tile staged via async global->LDS when available.
// EPI 0: f32 out = acc + bias
// EPI 1: f16 out = gelu(acc + bias)
// EPI 2: f32 out = acc + bias + add1 + add2
template<int EPI>
__global__ __launch_bounds__(256) void gemm_kernel(
    const f16* __restrict__ A, const f16* __restrict__ W,
    const float* __restrict__ bias,
    const float* __restrict__ add1, const float* __restrict__ add2,
    void* __restrict__ outp, int M, int N, int K)
{
  __shared__ f16 As[2][128 * 48];   // 128 rows (m), stride 48, K=32 valid
  __shared__ f16 Bs[2][128 * 48];   // 128 rows (n), stride 48, K=32 valid (W^T tile)
  int tid = threadIdx.x;
  int wid = tid >> 5, lane = tid & 31, lane16 = lane & 15, hi = lane >> 4;
  int wm = wid >> 2, wn = wid & 3;
  int bm = blockIdx.y * 128, bn = blockIdx.x * 128;

  // per-thread staging coordinates
  const int mA  = tid >> 1;          // A: row 0..127
  const int koA = (tid & 1) * 16;    // A: 16-half chunk within K=32
  const int krB = tid >> 3;          // W: k-row 0..31
  const int n0B = (tid & 7) * 16;    // W: 16-half chunk along n

  v8h wreg0, wreg1;

  auto issueA = [&](int buf, int kt){
    const f16* g = A + (size_t)(bm + mA) * K + kt + koA;
    f16* l = &As[buf][mA * 48 + koA];
#if HAVE_ASYNC_LDS
    __builtin_amdgcn_global_load_async_to_lds_b128(GLOBAL_V4(g),     LOCAL_V4(l),     0, 0);
    __builtin_amdgcn_global_load_async_to_lds_b128(GLOBAL_V4(g + 8), LOCAL_V4(l + 8), 0, 0);
#else
    v8h a0 = *(const v8h*)g;
    v8h a1 = *(const v8h*)(g + 8);
    *(v8h*)l       = a0;
    *(v8h*)(l + 8) = a1;
#endif
  };
  auto loadB = [&](int kt){
    const f16* g = W + (size_t)(kt + krB) * N + bn + n0B;
    wreg0 = *(const v8h*)g;
    wreg1 = *(const v8h*)(g + 8);
  };
  auto storeB = [&](int buf){
#pragma unroll
    for (int i = 0; i < 8; ++i){
      Bs[buf][(n0B + i) * 48 + krB]     = wreg0[i];
      Bs[buf][(n0B + 8 + i) * 48 + krB] = wreg1[i];
    }
  };

  F32x8 acc[4][2];
#pragma unroll
  for (int i = 0; i < 4; ++i)
#pragma unroll
    for (int j = 0; j < 2; ++j)
#pragma unroll
      for (int r = 0; r < 8; ++r) acc[i][j].f[r] = 0.0f;

  // prologue: stage tile 0
  issueA(0, 0);
  loadB(0);
  storeB(0);
#if HAVE_ASYNC_LDS
  WAIT_ASYNC0();
#endif
  __syncthreads();

  int buf = 0;
  for (int kt = 0; kt < K; kt += 32){
    bool more = (kt + 32) < K;
    if (more){ issueA(buf ^ 1, kt + 32); loadB(kt + 32); }

    F16x16 af[4], bf[2];
#pragma unroll
    for (int i = 0; i < 4; ++i){
      const f16* p = &As[buf][(wm*64 + i*16 + lane16) * 48 + hi*8];
      af[i].h[0] = *(const v8h*)p;
      af[i].h[1] = *(const v8h*)(p + 16);
    }
#pragma unroll
    for (int j = 0; j < 2; ++j){
      const f16* p = &Bs[buf][(wn*32 + j*16 + lane16) * 48 + hi*8];
      bf[j].h[0] = *(const v8h*)p;
      bf[j].h[1] = *(const v8h*)(p + 16);
    }
#pragma unroll
    for (int i = 0; i < 4; ++i)
#pragma unroll
      for (int j = 0; j < 2; ++j)
        acc[i][j].v = __builtin_amdgcn_wmma_f32_16x16x32_f16(
            false, af[i].v, false, bf[j].v, (short)0, acc[i][j].v, false, false);

    if (more){
      storeB(buf ^ 1);
#if HAVE_ASYNC_LDS
      WAIT_ASYNC0();
#endif
    }
    __syncthreads();
    buf ^= 1;
  }

  // epilogue
#pragma unroll
  for (int i = 0; i < 4; ++i){
#pragma unroll
    for (int j = 0; j < 2; ++j){
      int gn = bn + wn*32 + j*16 + lane16;
      float bv = bias[gn];
#pragma unroll
      for (int r = 0; r < 8; ++r){
        int gm = bm + wm*64 + i*16 + r + hi*8;
        float v = acc[i][j].f[r] + bv;
        size_t idx = (size_t)gm * N + gn;
        if (EPI == 1){
          ((f16*)outp)[idx] = (f16)gelu_exact(v);
        } else if (EPI == 2){
          ((float*)outp)[idx] = v + add1[idx] + add2[idx];
        } else {
          ((float*)outp)[idx] = v;
        }
      }
    }
  }
}

extern "C" void kernel_launch(void* const* d_in, const int* in_sizes, int n_in,
                              void* d_out, int out_size, void* d_ws, size_t ws_size,
                              hipStream_t stream) {
  const float* hidden = (const float*)d_in[0];
  const float* amask  = (const float*)d_in[1];
  const float* w_qkv  = (const float*)d_in[2];
  const float* b_qkv  = (const float*)d_in[3];
  const float* w_o    = (const float*)d_in[4];
  const float* b_o    = (const float*)d_in[5];
  const float* ln1s   = (const float*)d_in[6];
  const float* ln1b   = (const float*)d_in[7];
  const float* ln2s   = (const float*)d_in[8];
  const float* ln2b   = (const float*)d_in[9];
  const float* w_in   = (const float*)d_in[10];
  const float* b_in   = (const float*)d_in[11];
  const float* w_out  = (const float*)d_in[12];
  const float* b_out  = (const float*)d_in[13];

  char* ws = (char*)d_ws;
  const size_t MB = 1ull << 20;
  f16*   wqkv16 = (f16*)(ws + 0   * MB);   // 24 MB
  f16*   wo16   = (f16*)(ws + 24  * MB);   //  8 MB
  f16*   win16  = (f16*)(ws + 32  * MB);   // 32 MB
  f16*   wout16 = (f16*)(ws + 64  * MB);   // 32 MB
  f16*   x1     = (f16*)(ws + 96  * MB);   // 16 MB
  f16*   x2     = (f16*)(ws + 112 * MB);   // 16 MB
  f16*   q16    = (f16*)(ws + 128 * MB);   // 16 MB
  f16*   k16    = (f16*)(ws + 144 * MB);   // 16 MB
  f16*   vt16   = (f16*)(ws + 160 * MB);   // 16 MB
  f16*   ao16   = (f16*)(ws + 176 * MB);   // 16 MB
  float* qkvf   = (float*)(ws + 192 * MB); // 96 MB (dead after rope)
  f16*   hmid   = (f16*)(ws + 192 * MB);   // 64 MB (reuse)
  float* aproj  = (float*)(ws + 256 * MB); // 32 MB (reuse)

  cvt_f16_kernel<<<2048, 256, 0, stream>>>(w_qkv, wqkv16, HID * 3 * HID);
  cvt_f16_kernel<<<2048, 256, 0, stream>>>(w_o,   wo16,   HID * HID);
  cvt_f16_kernel<<<2048, 256, 0, stream>>>(w_in,  win16,  HID * INTER);
  cvt_f16_kernel<<<2048, 256, 0, stream>>>(w_out, wout16, INTER * HID);

  ln_dual_kernel<<<ROWS, 256, 0, stream>>>(hidden, ln1s, ln1b, ln2s, ln2b, x1, x2);

  gemm_kernel<0><<<dim3(3*HID/128, ROWS/128), 256, 0, stream>>>(
      x1, wqkv16, b_qkv, nullptr, nullptr, qkvf, ROWS, 3*HID, HID);

  rope_pack_kernel<<<ROWS, 256, 0, stream>>>(qkvf, q16, k16, vt16);

  attn_kernel<<<BATCH*HEADS*(SEQ/16)/8, 256, 0, stream>>>(q16, k16, vt16, amask, ao16);

  gemm_kernel<0><<<dim3(HID/128, ROWS/128), 256, 0, stream>>>(
      ao16, wo16, b_o, nullptr, nullptr, aproj, ROWS, HID, HID);

  gemm_kernel<1><<<dim3(INTER/128, ROWS/128), 256, 0, stream>>>(
      x2, win16, b_in, nullptr, nullptr, hmid, ROWS, INTER, HID);

  gemm_kernel<2><<<dim3(HID/128, ROWS/128), 256, 0, stream>>>(
      hmid, wout16, b_out, hidden, aproj, (float*)d_out, ROWS, HID, INTER);
}
